// AtomTransformerBlock_35150012351221
// MI455X (gfx1250) — matrix operand order repaired and checked by reading, not compiled, not run on hardware
//
#include <hip/hip_runtime.h>
#include <hip/hip_bf16.h>

#define NATOM 2048
#define CDIM  128
#define CP    16
#define HEADS 8
#define DDIM  16
#define WHALF 64
#define WSPAN 129          // window positions per query
#define EPSLN 1e-5f
#define NEGBIG -1e9f

// ---------------------------------------------------------------- types ----
typedef __bf16  bf16_t;
typedef __attribute__((ext_vector_type(16))) __bf16 v16bf;
typedef __attribute__((ext_vector_type(8)))  float  v8f;

union BF16Frag { v16bf v; unsigned short u[16]; uint4 q[2]; };
union F32Acc   { v8f   v; float f[8]; };

__device__ __forceinline__ unsigned short f32_to_bf16(float f) {
  unsigned int u = __float_as_uint(f);
  unsigned int r = u + 0x7FFFu + ((u >> 16) & 1u);   // round-to-nearest-even
  return (unsigned short)(r >> 16);
}
__device__ __forceinline__ float sigmoidf_(float x) {
  return 1.0f / (1.0f + __expf(-x));
}

// ------------------------------------------------------------- k_prep -----
// Per row i: LN(a) -> f32, LN(s)*adaln_scale -> bf16, LN(s)*t_scale -> bf16,
// s -> bf16.  blockDim = 128 (4 waves), grid = NATOM.
__global__ void k_prep(const float* __restrict__ a, const float* __restrict__ s,
                       const float* __restrict__ adaln_scale,
                       const float* __restrict__ t_scale,
                       float* __restrict__ ln_a,
                       unsigned short* __restrict__ sn_a_bf,
                       unsigned short* __restrict__ sn_t_bf,
                       unsigned short* __restrict__ s_bf) {
  const int i = blockIdx.x;
  const int c = threadIdx.x;               // 0..127
  const int idx = i * CDIM + c;
  const float av = a[idx], sv = s[idx];
  float r0 = av, r1 = av * av, r2 = sv, r3 = sv * sv;
#pragma unroll
  for (int off = 16; off > 0; off >>= 1) {
    r0 += __shfl_xor(r0, off, 32);
    r1 += __shfl_xor(r1, off, 32);
    r2 += __shfl_xor(r2, off, 32);
    r3 += __shfl_xor(r3, off, 32);
  }
  __shared__ float red[4][4];
  const int w = c >> 5;
  if ((c & 31) == 0) { red[w][0] = r0; red[w][1] = r1; red[w][2] = r2; red[w][3] = r3; }
  __syncthreads();
  const float S0 = red[0][0] + red[1][0] + red[2][0] + red[3][0];
  const float S1 = red[0][1] + red[1][1] + red[2][1] + red[3][1];
  const float S2 = red[0][2] + red[1][2] + red[2][2] + red[3][2];
  const float S3 = red[0][3] + red[1][3] + red[2][3] + red[3][3];
  const float invC = 1.0f / (float)CDIM;
  const float ma = S0 * invC, va = S1 * invC - ma * ma;
  const float ms = S2 * invC, vs = S3 * invC - ms * ms;
  const float ra = rsqrtf(va + EPSLN), rs = rsqrtf(vs + EPSLN);
  const float la = (av - ma) * ra;
  const float ls = (sv - ms) * rs;
  ln_a[idx]    = la;
  sn_a_bf[idx] = f32_to_bf16(ls * adaln_scale[c]);
  sn_t_bf[idx] = f32_to_bf16(ls * t_scale[c]);
  s_bf[idx]    = f32_to_bf16(sv);
}

// -------------------------------------------------------- WMMA GEMM -------
// C[M,N] = A_bf16[M,K] @ W_f32[K,N] (+ bias).
// blockDim = 256 (8 waves).  Block covers one 16-row M tile and a 128-column
// panel of N; each wave owns one 16x16 tile.  Weight panel is converted to
// bf16 cooperatively into LDS once per 128-deep K chunk (34 KB, padded row
// stride 136 halves -> 16-bit B reads of the two lane halves fall in different banks).  K and N are
// compile-time so the WMMA chain fully unrolls.
template<int K, int N>
__global__ void __launch_bounds__(256)
gemm_bf16_wmma(const unsigned short* __restrict__ A,
               const float* __restrict__ W,
               const float* __restrict__ bias,
               float* __restrict__ C, int M) {
  constexpr int KCHUNK = 128;
  constexpr int WROW   = 136;     // padded LDS row stride in halves
  __shared__ unsigned short lds_w[KCHUNK * WROW];

  constexpr int blkPerRow = N / 128;
  const int tm      = blockIdx.x / blkPerRow;
  const int colBase = (blockIdx.x % blkPerRow) * 128;
  const int tnl     = threadIdx.x >> 5;     // wave id = local 16-col tile
  const int lane    = threadIdx.x & 31;
  const int row     = lane & 15;
  const int hi      = lane >> 4;

  F32Acc acc;
#pragma unroll
  for (int r = 0; r < 8; ++r) acc.f[r] = 0.0f;

  const unsigned short* Arow = A + (size_t)(tm * 16 + row) * K;
  __builtin_prefetch(Arow, 0, 0);           // global_prefetch_b8

#pragma unroll
  for (int kc = 0; kc < K; kc += KCHUNK) {
    __syncthreads();                        // protect previous chunk's reads
    // cooperative stage: W[kc..kc+127][colBase..colBase+127] -> LDS (bf16)
    for (int e = threadIdx.x; e < KCHUNK * 64; e += 256) {
      const int kk = e >> 6;
      const int c2 = (e & 63) * 2;
      const float* wp = W + (size_t)(kc + kk) * N + colBase + c2;
      const unsigned int pack = (unsigned int)f32_to_bf16(wp[0]) |
                                ((unsigned int)f32_to_bf16(wp[1]) << 16);
      *(unsigned int*)(lds_w + kk * WROW + c2) = pack;
    }
    __syncthreads();
#pragma unroll
    for (int k0 = 0; k0 < KCHUNK; k0 += 32) {
      // A fragment (16x32 bf16): lane<16 row m=lane K{0..7,16..23},
      // lane>=16 row m=lane-16 K{8..15,24..31}   (CDNA5 ISA 7.12.2)
      BF16Frag af;
      af.q[0] = *(const uint4*)(Arow + kc + k0 + hi * 8);
      af.q[1] = *(const uint4*)(Arow + kc + k0 + 16 + hi * 8);
      // B fragment (32x16 bf16): lane = N column, K order same as A
      BF16Frag bfr;
      const unsigned short* wr = lds_w + tnl * 16 + row;
#pragma unroll
      for (int s = 0; s < 16; ++s) {
        const int kk = k0 + (s < 8 ? hi * 8 + s : 16 + hi * 8 + (s - 8));
        bfr.u[s] = wr[kk * WROW];
      }
      acc.v = __builtin_amdgcn_wmma_f32_16x16x32_bf16(
          false, af.v, false, bfr.v, (short)0, acc.v, false, false);
    }
  }
  // C/D layout: VGPR r -> (M=r, N=lane) lanes 0-15 ; (M=r+8, N=lane-16)
  const int col = colBase + tnl * 16 + row;
  const float bv = bias ? bias[col] : 0.0f;
#pragma unroll
  for (int r = 0; r < 8; ++r) {
    const int m = tm * 16 + r + hi * 8;
    C[(size_t)m * N + col] = acc.f[r] + bv;
  }
}

// --------------------------------------------------- AdaLN combine --------
// out_bf = bf16( sigmoid(gate) * ln_a + skip )
__global__ void k_adaln_combine(const float* __restrict__ gate,
                                const float* __restrict__ skip,
                                const float* __restrict__ ln_a,
                                unsigned short* __restrict__ out_bf, int total) {
  const int t = blockIdx.x * blockDim.x + threadIdx.x;
  if (t < total)
    out_bf[t] = f32_to_bf16(sigmoidf_(gate[t]) * ln_a[t] + skip[t]);
}

// ------------------------------------------------------- pair bias --------
// bias[i][jj][h] = LN(pair[i,j,:])@w_pair[:,h] + 1e9*(mask[j]-1), window only
__global__ void k_pairbias(const float* __restrict__ pair,
                           const float* __restrict__ ln_scale,
                           const float* __restrict__ ln_bias,
                           const float* __restrict__ w_pair,   // [CP,H]
                           const float* __restrict__ mask,
                           float* __restrict__ bias_out) {
  const int t = blockIdx.x * blockDim.x + threadIdx.x;
  if (t >= NATOM * WSPAN) return;
  const int i  = t / WSPAN;
  const int jj = t % WSPAN;
  const int j  = i - WHALF + jj;
  float out[HEADS];
  if (j < 0 || j >= NATOM) {
#pragma unroll
    for (int h = 0; h < HEADS; ++h) out[h] = NEGBIG;
  } else {
    const float* p = pair + ((size_t)i * NATOM + (size_t)j) * CP;
    float ln[CP];
    float m = 0.0f;
#pragma unroll
    for (int c = 0; c < CP; ++c) { ln[c] = p[c]; m += ln[c]; }
    m *= (1.0f / CP);
    float v = 0.0f;
#pragma unroll
    for (int c = 0; c < CP; ++c) { const float d = ln[c] - m; v += d * d; }
    v *= (1.0f / CP);
    const float r = rsqrtf(v + EPSLN);
#pragma unroll
    for (int c = 0; c < CP; ++c) ln[c] = (ln[c] - m) * r * ln_scale[c] + ln_bias[c];
    const float mb = 1e9f * (mask[j] - 1.0f);
#pragma unroll
    for (int h = 0; h < HEADS; ++h) {
      float acc = 0.0f;
#pragma unroll
      for (int c = 0; c < CP; ++c) acc += ln[c] * w_pair[c * HEADS + h];
      out[h] = acc + mb;
    }
  }
  float* dst = bias_out + (size_t)t * HEADS;
#pragma unroll
  for (int h = 0; h < HEADS; ++h) dst[h] = out[h];
}

// ------------------------------------------------------- attention --------
// grid = NATOM blocks, blockDim = 256: wave = head.  Two-pass windowed
// softmax, per-head sigmoid output gating; writes og (bf16) for wo GEMM.
__global__ void k_attn(const float* __restrict__ q, const float* __restrict__ k,
                       const float* __restrict__ v, const float* __restrict__ gq,
                       const float* __restrict__ bias,
                       unsigned short* __restrict__ og_bf) {
  const int i    = blockIdx.x;
  const int h    = threadIdx.x >> 5;
  const int lane = threadIdx.x & 31;

  float qv[DDIM];
  const float* qp = q + (size_t)i * CDIM + h * DDIM;
#pragma unroll
  for (int d = 0; d < DDIM; ++d) qv[d] = qp[d] * 0.25f;   // D^-0.5 = 1/4

  // pass 1: scores for jj = lane + 32*t
  float sc[5];
  float mx = -1e30f;
#pragma unroll
  for (int t = 0; t < 5; ++t) {
    const int jj = lane + t * 32;
    float s = -1e30f;
    if (jj < WSPAN) {
      const int j = i - WHALF + jj;
      if (j >= 0 && j < NATOM) {
        const float* kp = k + (size_t)j * CDIM + h * DDIM;
        float acc = 0.0f;
#pragma unroll
        for (int d = 0; d < DDIM; ++d) acc += qv[d] * kp[d];
        s = acc + bias[((size_t)i * WSPAN + jj) * HEADS + h];
      }
    }
    sc[t] = s;
    mx = fmaxf(mx, s);
  }
#pragma unroll
  for (int off = 16; off > 0; off >>= 1) mx = fmaxf(mx, __shfl_xor(mx, off, 32));
  float sum = 0.0f;
#pragma unroll
  for (int t = 0; t < 5; ++t) {
    const float e = (sc[t] > -1e29f) ? __expf(sc[t] - mx) : 0.0f;
    sc[t] = e;
    sum += e;
  }
#pragma unroll
  for (int off = 16; off > 0; off >>= 1) sum += __shfl_xor(sum, off, 32);
  const float inv = 1.0f / sum;

  // pass 2: weighted V accumulate
  float o[DDIM];
#pragma unroll
  for (int d = 0; d < DDIM; ++d) o[d] = 0.0f;
#pragma unroll
  for (int t = 0; t < 5; ++t) {
    const int jj = lane + t * 32;
    if (jj < WSPAN && sc[t] > 0.0f) {
      const int j = i - WHALF + jj;
      const float* vp = v + (size_t)j * CDIM + h * DDIM;
#pragma unroll
      for (int d = 0; d < DDIM; ++d) o[d] += sc[t] * vp[d];
    }
  }
#pragma unroll
  for (int off = 16; off > 0; off >>= 1) {
#pragma unroll
    for (int d = 0; d < DDIM; ++d) o[d] += __shfl_xor(o[d], off, 32);
  }
  if (lane < DDIM) {
    float mine = 0.0f;
#pragma unroll
    for (int d = 0; d < DDIM; ++d) mine = (lane == d) ? o[d] : mine;
    const int c = h * DDIM + lane;
    const float gg = sigmoidf_(gq[(size_t)i * CDIM + c]);
    og_bf[(size_t)i * CDIM + c] = f32_to_bf16(mine * inv * gg);
  }
}

// --------------------------------------------------------- SwiGLU ---------
__global__ void k_swiglu(const float* __restrict__ h1, const float* __restrict__ h2,
                         unsigned short* __restrict__ hid_bf, int total) {
  const int t = blockIdx.x * blockDim.x + threadIdx.x;
  if (t < total) {
    const float x = h1[t];
    hid_bf[t] = f32_to_bf16(x * sigmoidf_(x) * h2[t]);
  }
}

// ------------------------------------------------------- final sum --------
__global__ void k_final(const float* __restrict__ g1, const float* __restrict__ ao,
                        const float* __restrict__ g2, const float* __restrict__ tp,
                        float* __restrict__ out, int total) {
  const int t = blockIdx.x * blockDim.x + threadIdx.x;
  if (t < total)
    out[t] = sigmoidf_(g1[t]) * ao[t] + sigmoidf_(g2[t]) * tp[t];
}

// ---------------------------------------------------------------- host ----
extern "C" void kernel_launch(void* const* d_in, const int* in_sizes, int n_in,
                              void* d_out, int out_size, void* d_ws, size_t ws_size,
                              hipStream_t stream) {
  const float* atom_single  = (const float*)d_in[0];
  const float* atom_proj    = (const float*)d_in[1];
  const float* pair         = (const float*)d_in[2];
  const float* mask         = (const float*)d_in[3];
  const float* adaln_scale  = (const float*)d_in[4];
  const float* adaln_gate_w = (const float*)d_in[5];
  const float* adaln_gate_b = (const float*)d_in[6];
  const float* adaln_skip_w = (const float*)d_in[7];
  const float* wq           = (const float*)d_in[8];
  const float* wk           = (const float*)d_in[9];
  const float* wv           = (const float*)d_in[10];
  const float* wg           = (const float*)d_in[11];
  const float* bg           = (const float*)d_in[12];
  const float* wo           = (const float*)d_in[13];
  const float* bo           = (const float*)d_in[14];
  const float* pair_ln_s    = (const float*)d_in[15];
  const float* pair_ln_b    = (const float*)d_in[16];
  const float* w_pair       = (const float*)d_in[17];
  const float* out_gate_w   = (const float*)d_in[18];
  const float* out_gate_b   = (const float*)d_in[19];
  const float* t_s_scale    = (const float*)d_in[20];
  const float* t_gate_w     = (const float*)d_in[21];
  const float* t_gate_b     = (const float*)d_in[22];
  const float* t_skip_w     = (const float*)d_in[23];
  const float* t_swish_w    = (const float*)d_in[24];
  const float* t_hidden_w   = (const float*)d_in[25];
  const float* t_out_w      = (const float*)d_in[26];
  const float* t_out_gate_w = (const float*)d_in[27];
  const float* t_out_gate_b = (const float*)d_in[28];
  (void)in_sizes; (void)n_in; (void)out_size; (void)ws_size;

  // -------- workspace carve-up (256B aligned) --------
  char* base = (char*)d_ws;
  size_t off = 0;
  auto carve = [&](size_t bytes) -> char* {
    char* p = base + off;
    off += (bytes + 255) & ~(size_t)255;
    return p;
  };
  const size_t NC  = (size_t)NATOM * CDIM;
  const size_t NC2 = (size_t)NATOM * 2 * CDIM;

  float*          ln_a    = (float*)carve(NC * 4);
  unsigned short* sn_a_bf = (unsigned short*)carve(NC * 2);
  unsigned short* sn_t_bf = (unsigned short*)carve(NC * 2);
  unsigned short* s_bf    = (unsigned short*)carve(NC * 2);
  unsigned short* a_bf    = (unsigned short*)carve(NC * 2);   // reused as t_bf
  unsigned short* og_bf   = (unsigned short*)carve(NC * 2);
  unsigned short* hid_bf  = (unsigned short*)carve(NC2 * 2);
  float* ws_gate = (float*)carve(NC * 4);                     // ga, then tg
  float* ws_skip = (float*)carve(NC * 4);                     // sk, then tsk
  float* ws_q    = (float*)carve(NC * 4);                     // q, then trans_pre
  float* ws_k    = (float*)carve(NC * 4);                     // k, then gate2
  float* ws_v    = (float*)carve(NC * 4);
  float* ws_gq   = (float*)carve(NC * 4);
  float* ws_ao   = (float*)carve(NC * 4);
  float* ws_g1   = (float*)carve(NC * 4);
  float* ws_h1   = (float*)carve(NC2 * 4);
  float* ws_h2   = (float*)carve(NC2 * 4);
  float* ws_bias = (float*)carve((size_t)NATOM * WSPAN * HEADS * 4);

  // GEMM grid: one block per (16-row M tile) x (128-col N panel)
#define GEMM(KK, NN, Aptr, Wptr, Bptr, Cptr)                                  \
  gemm_bf16_wmma<KK, NN><<<(NATOM / 16) * ((NN) / 128), 256, 0, stream>>>(    \
      Aptr, Wptr, Bptr, Cptr, NATOM)

  // 1) layernorms / bf16 staging
  k_prep<<<NATOM, CDIM, 0, stream>>>(atom_single, atom_proj, adaln_scale,
                                     t_s_scale, ln_a, sn_a_bf, sn_t_bf, s_bf);
  // 2) pair bias (windowed only: 2048x129 instead of 2048x2048)
  {
    const int total = NATOM * WSPAN;
    k_pairbias<<<(total + 127) / 128, 128, 0, stream>>>(pair, pair_ln_s, pair_ln_b,
                                                        w_pair, mask, ws_bias);
  }
  // 3) attention AdaLN: sigmoid(sn@Wg + b)*LN(a) + sn@Ws
  GEMM(128, 128, sn_a_bf, adaln_gate_w, adaln_gate_b, ws_gate);
  GEMM(128, 128, sn_a_bf, adaln_skip_w, nullptr,      ws_skip);
  k_adaln_combine<<<(int)(NC / 256), 256, 0, stream>>>(ws_gate, ws_skip, ln_a,
                                                       a_bf, (int)NC);
  // 4) Q K V G projections
  GEMM(128, 128, a_bf, wq, nullptr, ws_q);
  GEMM(128, 128, a_bf, wk, nullptr, ws_k);
  GEMM(128, 128, a_bf, wv, nullptr, ws_v);
  GEMM(128, 128, a_bf, wg, bg,      ws_gq);
  // 5) windowed attention + per-head gate
  k_attn<<<NATOM, 256, 0, stream>>>(ws_q, ws_k, ws_v, ws_gq, ws_bias, og_bf);
  // 6) output projection + adaLN-zero gate
  GEMM(128, 128, og_bf, wo, bo, ws_ao);
  GEMM(128, 128, s_bf, out_gate_w, out_gate_b, ws_g1);
  // 7) transition AdaLN (reuse gate/skip buffers; a_bf becomes t_bf)
  GEMM(128, 128, sn_t_bf, t_gate_w, t_gate_b, ws_gate);
  GEMM(128, 128, sn_t_bf, t_skip_w, nullptr,  ws_skip);
  k_adaln_combine<<<(int)(NC / 256), 256, 0, stream>>>(ws_gate, ws_skip, ln_a,
                                                       a_bf, (int)NC);
  // 8) SwiGLU transition (expansion 2)
  GEMM(128, 256, a_bf, t_swish_w,  nullptr, ws_h1);
  GEMM(128, 256, a_bf, t_hidden_w, nullptr, ws_h2);
  k_swiglu<<<(int)(NC2 / 256), 256, 0, stream>>>(ws_h1, ws_h2, hid_bf, (int)NC2);
  GEMM(256, 128, hid_bf, t_out_w, nullptr, ws_q);            // trans_pre
  GEMM(128, 128, s_bf, t_out_gate_w, t_out_gate_b, ws_k);    // gate2
  // 9) final: sigmoid(g1)*attn_out + sigmoid(g2)*trans
  k_final<<<(int)(NC / 256), 256, 0, stream>>>(ws_g1, ws_ao, ws_k, ws_q,
                                               (float*)d_out, (int)NC);
#undef GEMM
}
